// Vully_9320079033241
// MI455X (gfx1250) — matrix operand order repaired and checked by reading, not compiled, not run on hardware
//
#include <hip/hip_runtime.h>
#include <hip/hip_bf16.h>
#include <stdint.h>

typedef __attribute__((ext_vector_type(16))) _Float16 v16h;
typedef __attribute__((ext_vector_type(8)))  _Float16 v8h;
typedef __attribute__((ext_vector_type(8)))  float    v8f;

#define N_NODES 10000
#define N_EDGES 80000
#define BATCHSZ 64
#define SEQT    512
#define EMBD    100
#define HEAD1   2
#define CH1     1000
#define HEAD2   2
#define CH2     500
#define GRUH    100

static inline size_t rup(size_t v, size_t m) { return (v + m - 1) / m * m; }

__device__ __forceinline__ float elu1(float x)   { return x > 0.f ? x : (__expf(x) - 1.f); }
__device__ __forceinline__ float leaky02(float x){ return x > 0.f ? x : 0.2f * x; }
__device__ __forceinline__ float sigm(float x)   { return 1.f / (1.f + __expf(-x)); }

__device__ __forceinline__ void atomicMaxF(float* addr, float val) {
  int* ai = reinterpret_cast<int*>(addr);
  int old = *ai;
  while (__int_as_float(old) < val) {
    int assumed = old;
    old = atomicCAS(ai, assumed, __float_as_int(val));
    if (old == assumed) break;
  }
}

// ---------------------------------------------------------------------------
// WMMA GEMM. A: [M x Kp] f16 row-major (Kp mult 32, M mult 16, zero padded).
// Bsw: weights pre-swizzled into WMMA B-fragment order (see pack_bfrag_kernel):
//   element(((tn*nKc + kc)*32 + lane)*16 + j) = B[kc*32 + (lane>>4)*16 + j][tn*16 + (lane&15)]
// One wave computes TWO 16x16 C tiles (shared A fragment).
// grid = (Np/32, M/16), blockDim = 32. Np multiple of 32.
// ---------------------------------------------------------------------------
template<int ACT, bool HASBIAS>
__global__ void wmma_gemm_kernel(const _Float16* __restrict__ A,
                                 const _Float16* __restrict__ Bsw,
                                 float* __restrict__ C,
                                 const float* __restrict__ bias,
                                 int M, int N, int Kp, int Np, int ldc)
{
  const int lane = threadIdx.x;
  const int half = lane >> 4;
  const int lq   = lane & 15;
  const int row  = blockIdx.y * 16 + lq;     // always < M
  const int nKc  = Kp >> 5;
  const int tn0  = blockIdx.x * 2;

  v8f acc0 = {}, acc1 = {};
  const _Float16* Ap  = A + (size_t)row * Kp + half * 16;
  const _Float16* Bp0 = Bsw + ((size_t)tn0 * nKc * 32 + lane) * 16;
  const _Float16* Bp1 = Bp0 + (size_t)nKc * 512;

  for (int kc = 0; kc < nKc; ++kc) {
    const int k0 = kc * 32;
    __builtin_prefetch(Ap + k0 + 32, 0, 1);          // global_prefetch on A stream
    // A fragment: contiguous 32B per lane (k = 16*half + s), matching B packing
    v8h aLo = *(const v8h*)(Ap + k0);
    v8h aHi = *(const v8h*)(Ap + k0 + 8);
    v16h a = __builtin_shufflevector(aLo, aHi, 0,1,2,3,4,5,6,7,8,9,10,11,12,13,14,15);
    // B fragments: contiguous 32B per lane in the swizzled buffer
    v8h b0lo = *(const v8h*)(Bp0 + (size_t)kc * 512);
    v8h b0hi = *(const v8h*)(Bp0 + (size_t)kc * 512 + 8);
    v16h b0 = __builtin_shufflevector(b0lo, b0hi, 0,1,2,3,4,5,6,7,8,9,10,11,12,13,14,15);
    acc0 = __builtin_amdgcn_wmma_f32_16x16x32_f16(false, a, false, b0, (short)0, acc0, false, false);
    v8h b1lo = *(const v8h*)(Bp1 + (size_t)kc * 512);
    v8h b1hi = *(const v8h*)(Bp1 + (size_t)kc * 512 + 8);
    v16h b1 = __builtin_shufflevector(b1lo, b1hi, 0,1,2,3,4,5,6,7,8,9,10,11,12,13,14,15);
    acc1 = __builtin_amdgcn_wmma_f32_16x16x32_f16(false, a, false, b1, (short)0, acc1, false, false);
  }

  const int col0 = blockIdx.x * 32 + lq;
  const int col1 = col0 + 16;
  const float bv0 = (HASBIAS && col0 < N) ? bias[col0] : 0.f;
  const float bv1 = (HASBIAS && col1 < N) ? bias[col1] : 0.f;
#pragma unroll
  for (int v = 0; v < 8; ++v) {
    const int crow = blockIdx.y * 16 + half * 8 + v;  // always < M
    if (col0 < N) {
      float x = acc0[v] + bv0;
      if (ACT == 1) x = fmaxf(x, 0.f);
      C[(size_t)crow * ldc + col0] = x;
    }
    if (col1 < N) {
      float x = acc1[v] + bv1;
      if (ACT == 1) x = fmaxf(x, 0.f);
      C[(size_t)crow * ldc + col1] = x;
    }
  }
}

// ---------------------------------------------------------------------------
// Pack weights into WMMA B-fragment order (f32 -> f16, zero padded).
// Logical B is [K x N]; trans=1 means src is stored [N x K] (e.g. GRU Wih).
// ---------------------------------------------------------------------------
__global__ void pack_bfrag_kernel(const float* __restrict__ src, _Float16* __restrict__ dst,
                                  int K, int N, int Kp, int Np, int trans)
{
  size_t i = (size_t)blockIdx.x * blockDim.x + threadIdx.x;
  size_t tot = (size_t)Np * Kp;
  if (i >= tot) return;
  const int nKc = Kp >> 5;
  int j    = (int)(i & 15);
  int lane = (int)((i >> 4) & 31);
  size_t blk = i >> 9;
  int tn = (int)(blk / nKc);
  int kc = (int)(blk % nKc);
  int k = kc * 32 + (lane >> 4) * 16 + j;
  int n = tn * 16 + (lane & 15);
  float v = 0.f;
  if (k < K && n < N)
    v = trans ? src[(size_t)n * K + k] : src[(size_t)k * N + n];
  dst[i] = (_Float16)v;
}

// ---------------------------------------------------------------------------
// A-side packing / conversion kernels
// ---------------------------------------------------------------------------
__global__ void pack_f16_kernel(const float* __restrict__ src, _Float16* __restrict__ dst,
                                int R, int Cc, int Cp)
{
  size_t i = (size_t)blockIdx.x * blockDim.x + threadIdx.x;
  size_t tot = (size_t)R * Cp;
  if (i >= tot) return;
  int c = (int)(i % Cp);
  int r = (int)(i / Cp);
  float v = (c < Cc) ? src[(size_t)r * Cc + c] : 0.f;
  dst[i] = (_Float16)v;
}

__global__ void gather_pack_kernel(const int* __restrict__ tok, const float* __restrict__ emb,
                                   _Float16* __restrict__ dst, int Rows, int K, int Cp)
{
  size_t i = (size_t)blockIdx.x * blockDim.x + threadIdx.x;
  size_t tot = (size_t)Rows * Cp;
  if (i >= tot) return;
  int r = (int)(i / Cp), c = (int)(i % Cp);
  float v = (c < K) ? emb[(size_t)tok[r] * K + c] : 0.f;
  dst[i] = (_Float16)v;
}

__global__ void fill_f32_kernel(float* __restrict__ p, float v, size_t n)
{
  size_t i = (size_t)blockIdx.x * blockDim.x + threadIdx.x;
  if (i < n) p[i] = v;
}

// ---------------------------------------------------------------------------
// GAT attention kernels
// ---------------------------------------------------------------------------
__global__ void attn_node_kernel(const float* __restrict__ XW,
                                 const float* __restrict__ a_src,
                                 const float* __restrict__ a_dst,
                                 float* __restrict__ asrc, float* __restrict__ adst,
                                 int Nn, int H, int C)
{
  int i = blockIdx.x * blockDim.x + threadIdx.x;
  if (i >= Nn * H) return;
  int n = i / H, h = i % H;
  const float* x = XW + (size_t)n * H * C + (size_t)h * C;
  float s1 = 0.f, s2 = 0.f;
  for (int c = 0; c < C; ++c) {
    float xv = x[c];
    s1 += xv * a_src[h * C + c];
    s2 += xv * a_dst[h * C + c];
  }
  asrc[i] = s1;
  adst[i] = s2;
}

__global__ void edge_max_kernel(const int* __restrict__ ei, int E, int Nn, int H,
                                const float* __restrict__ asrc, const float* __restrict__ adst,
                                float* __restrict__ m)
{
  int e = blockIdx.x * blockDim.x + threadIdx.x;
  int Et = E + Nn;
  if (e >= Et) return;
  int s = (e < E) ? ei[e] : (e - E);
  int d = (e < E) ? ei[E + e] : (e - E);
  for (int h = 0; h < H; ++h)
    atomicMaxF(&m[d * H + h], leaky02(asrc[s * H + h] + adst[d * H + h]));
}

__global__ void edge_denom_kernel(const int* __restrict__ ei, int E, int Nn, int H,
                                  const float* __restrict__ asrc, const float* __restrict__ adst,
                                  const float* __restrict__ m, float* __restrict__ den)
{
  int e = blockIdx.x * blockDim.x + threadIdx.x;
  int Et = E + Nn;
  if (e >= Et) return;
  int s = (e < E) ? ei[e] : (e - E);
  int d = (e < E) ? ei[E + e] : (e - E);
  for (int h = 0; h < H; ++h) {
    float l = leaky02(asrc[s * H + h] + adst[d * H + h]);
    atomicAdd(&den[d * H + h], __expf(l - m[d * H + h]));
  }
}

__global__ void edge_scatter_kernel(const int* __restrict__ ei, int E, int Nn, int H, int C,
                                    const float* __restrict__ asrc, const float* __restrict__ adst,
                                    const float* __restrict__ m, const float* __restrict__ den,
                                    const float* __restrict__ XW, float* __restrict__ OUT)
{
  size_t i = (size_t)blockIdx.x * blockDim.x + threadIdx.x;
  int HC = H * C;
  size_t tot = (size_t)(E + Nn) * HC;
  if (i >= tot) return;
  int e = (int)(i / HC);
  int f = (int)(i % HC);
  int h = f / C;
  int s = (e < E) ? ei[e] : (e - E);
  int d = (e < E) ? ei[E + e] : (e - E);
  float l = leaky02(asrc[s * H + h] + adst[d * H + h]);
  float ex = __expf(l - m[d * H + h]);
  float att = ex / (den[d * H + h] + 1e-16f);
  atomicAdd(&OUT[(size_t)d * HC + f], XW[(size_t)s * HC + f] * att);
}

__global__ void bias_elu_kernel(float* __restrict__ x, const float* __restrict__ bias,
                                int Nn, int F)
{
  size_t i = (size_t)blockIdx.x * blockDim.x + threadIdx.x;
  size_t tot = (size_t)Nn * F;
  if (i >= tot) return;
  x[i] = elu1(x[i] + bias[i % F]);
}

// ---------------------------------------------------------------------------
// global_max_pool
// ---------------------------------------------------------------------------
__global__ void pool_max_kernel(const float* __restrict__ x, const int* __restrict__ batch,
                                float* __restrict__ g, int Nn, int F)
{
  size_t i = (size_t)blockIdx.x * blockDim.x + threadIdx.x;
  size_t tot = (size_t)Nn * F;
  if (i >= tot) return;
  int n = (int)(i / F), f = (int)(i % F);
  atomicMaxF(&g[(size_t)batch[n] * F + f], x[i]);
}

__global__ void pool_fix_kernel(float* __restrict__ g, size_t n)
{
  size_t i = (size_t)blockIdx.x * blockDim.x + threadIdx.x;
  if (i < n) g[i] = (g[i] < -1e29f) ? 0.f : g[i];
}

// ---------------------------------------------------------------------------
// GRU recurrence: one workgroup per batch element, h in LDS.
// ---------------------------------------------------------------------------
__global__ void gru_scan_kernel(const float* __restrict__ gi,
                                const float* __restrict__ Whh,
                                const float* __restrict__ bhh,
                                float* __restrict__ ys,
                                float* __restrict__ hT)
{
  __shared__ float h[GRUH];
  __shared__ float gh[3 * GRUH];
  const int b = blockIdx.x;
  const int j = threadIdx.x;  // blockDim = 320
  if (j < GRUH) h[j] = 0.f;
  __syncthreads();
  for (int t = 0; t < SEQT; ++t) {
    if (j < 3 * GRUH) {
      float acc = bhh[j];
      const float* w = Whh + j * GRUH;
#pragma unroll 4
      for (int k = 0; k < GRUH; ++k) acc += h[k] * w[k];
      gh[j] = acc;
    }
    __syncthreads();
    float hn = 0.f;
    const bool on = (j < GRUH);
    if (on) {
      const float* gib = gi + ((size_t)b * SEQT + t) * (3 * GRUH);
      float r  = sigm(gib[j]             + gh[j]);
      float z  = sigm(gib[GRUH + j]      + gh[GRUH + j]);
      float nn = tanhf(gib[2 * GRUH + j] + r * gh[2 * GRUH + j]);
      hn = (1.f - z) * nn + z * h[j];
    }
    __syncthreads();
    if (on) {
      h[j] = hn;
      ys[((size_t)b * SEQT + t) * GRUH + j] = hn;
    }
    __syncthreads();
  }
  if (j < GRUH) hT[b * GRUH + j] = h[j];
}

// ---------------------------------------------------------------------------
// Head concat: z[b,0:1000]=g, z[b,1000:1500]=[h0,h1,h0,h1,h2], pad to 1504
// ---------------------------------------------------------------------------
__global__ void concat_head_kernel(const float* __restrict__ g, const float* __restrict__ hT,
                                   _Float16* __restrict__ z)
{
  int i = blockIdx.x * blockDim.x + threadIdx.x;
  const int ZC = 1504;
  if (i >= BATCHSZ * ZC) return;
  int b = i / ZC, c = i % ZC;
  float v = 0.f;
  if (c < 1000) {
    v = g[(size_t)b * 1000 + c];
  } else if (c < 1500) {
    int jj = c - 1000;
    int seg = jj / 100, u = jj % 100;
    int layer = (seg == 0) ? 0 : (seg == 1) ? 1 : (seg == 2) ? 0 : (seg == 3) ? 1 : 2;
    v = hT[layer * (BATCHSZ * GRUH) + b * GRUH + u];
  }
  z[i] = (_Float16)v;
}

// ===========================================================================
extern "C" void kernel_launch(void* const* d_in, const int* in_sizes, int n_in,
                              void* d_out, int out_size, void* d_ws, size_t ws_size,
                              hipStream_t stream)
{
  (void)in_sizes; (void)n_in; (void)out_size; (void)ws_size;

  const int*   node_tokens = (const int*)d_in[0];
  const int*   edge_index  = (const int*)d_in[1];
  const int*   batch       = (const int*)d_in[2];
  const int*   tokens      = (const int*)d_in[3];
  const float* emb1        = (const float*)d_in[4];
  const float* gat1_W      = (const float*)d_in[5];
  const float* gat1_as     = (const float*)d_in[6];
  const float* gat1_ad     = (const float*)d_in[7];
  const float* gat1_b      = (const float*)d_in[8];
  const float* gat2_W      = (const float*)d_in[9];
  const float* gat2_as     = (const float*)d_in[10];
  const float* gat2_ad     = (const float*)d_in[11];
  const float* gat2_b      = (const float*)d_in[12];
  const float* emb2        = (const float*)d_in[13];
  const float* lin1_W      = (const float*)d_in[14];
  const float* lin1_b      = (const float*)d_in[15];
  const float* lin11_W     = (const float*)d_in[16];
  const float* lin11_b     = (const float*)d_in[17];
  const float* lin2_W      = (const float*)d_in[18];
  const float* lin2_b      = (const float*)d_in[19];
  const float* Wih[3] = {(const float*)d_in[20], (const float*)d_in[24], (const float*)d_in[28]};
  const float* Whh[3] = {(const float*)d_in[21], (const float*)d_in[25], (const float*)d_in[29]};
  const float* bih[3] = {(const float*)d_in[22], (const float*)d_in[26], (const float*)d_in[30]};
  const float* bhh[3] = {(const float*)d_in[23], (const float*)d_in[27], (const float*)d_in[31]};
  float* out = (float*)d_out;

  // ---- workspace bump allocator ------------------------------------------
  char* base = (char*)d_ws;
  size_t off = 0;
  auto alloc = [&](size_t bytes) -> void* {
    void* p = base + off;
    off += rup(bytes, 256);
    return p;
  };

  const int BT  = BATCHSZ * SEQT;   // 32768
  const int Kp0 = 128;              // 100 -> mult 32
  const int Kp1 = 2016;             // 2000 -> mult 32
  const int Np1 = 2016;             // GAT1 out cols, mult 32
  const int Np2 = 1024;             // GAT2 out cols, mult 32
  const int NpG = 320;              // GRU gates 300 -> mult 32

  _Float16* a0f16  = (_Float16*)alloc((size_t)N_NODES * Kp0 * 2);
  _Float16* w1f16  = (_Float16*)alloc((size_t)Kp0 * Np1 * 2);
  float*    xw1    = (float*)alloc((size_t)N_NODES * 2000 * 4);   // aliased later as OUT2
  float*    asrc1  = (float*)alloc((size_t)N_NODES * HEAD1 * 4);
  float*    adst1  = (float*)alloc((size_t)N_NODES * HEAD1 * 4);
  float*    m1     = (float*)alloc((size_t)N_NODES * HEAD1 * 4);
  float*    den1   = (float*)alloc((size_t)N_NODES * HEAD1 * 4);
  float*    out1   = (float*)alloc((size_t)N_NODES * 2000 * 4);   // aliased later as XW2
  _Float16* x1f16  = (_Float16*)alloc((size_t)N_NODES * Kp1 * 2);
  _Float16* w2f16  = (_Float16*)alloc((size_t)Kp1 * Np2 * 2);
  float*    asrc2  = (float*)alloc((size_t)N_NODES * HEAD2 * 4);
  float*    adst2  = (float*)alloc((size_t)N_NODES * HEAD2 * 4);
  float*    m2     = (float*)alloc((size_t)N_NODES * HEAD2 * 4);
  float*    den2   = (float*)alloc((size_t)N_NODES * HEAD2 * 4);
  float*    gpool  = (float*)alloc((size_t)BATCHSZ * 1000 * 4);
  _Float16* temb   = (_Float16*)alloc((size_t)BT * Kp0 * 2);
  _Float16* wihT[3];
  for (int l = 0; l < 3; ++l) wihT[l] = (_Float16*)alloc((size_t)Kp0 * NpG * 2);
  float*    gibuf  = (float*)alloc((size_t)BT * 300 * 4);
  float*    ysA    = (float*)alloc((size_t)BT * GRUH * 4);
  _Float16* ysf16  = (_Float16*)alloc((size_t)BT * Kp0 * 2);
  float*    hTbuf  = (float*)alloc((size_t)3 * BATCHSZ * GRUH * 4);
  _Float16* zcat   = (_Float16*)alloc((size_t)BATCHSZ * 1504 * 2);
  _Float16* wl1    = (_Float16*)alloc((size_t)1504 * 1024 * 2);
  float*    z1     = (float*)alloc((size_t)BATCHSZ * 1000 * 4);
  _Float16* z1f16  = (_Float16*)alloc((size_t)BATCHSZ * 1024 * 2);
  _Float16* wl11   = (_Float16*)alloc((size_t)1024 * 512 * 2);
  float*    z2     = (float*)alloc((size_t)BATCHSZ * 500 * 4);
  _Float16* z2f16  = (_Float16*)alloc((size_t)BATCHSZ * 512 * 2);
  _Float16* wl2    = (_Float16*)alloc((size_t)512 * 32 * 2);

  float* xw2  = out1;   // alias (out1 fp32 dead after packing into x1f16)
  float* out2 = xw1;    // alias (xw1 dead after GAT1 scatter)

  auto gemm = [&](const _Float16* A, const _Float16* B, float* C, const float* bias,
                  int M, int N, int Kp, int Np, int ldc, int act) {
    dim3 grid(Np / 32, M / 16);
    if (bias) {
      if (act == 1) wmma_gemm_kernel<1, true><<<grid, 32, 0, stream>>>(A, B, C, bias, M, N, Kp, Np, ldc);
      else          wmma_gemm_kernel<0, true><<<grid, 32, 0, stream>>>(A, B, C, bias, M, N, Kp, Np, ldc);
    } else {
      wmma_gemm_kernel<0, false><<<grid, 32, 0, stream>>>(A, B, C, nullptr, M, N, Kp, Np, ldc);
    }
  };
  auto nblk = [](size_t tot, int bs) { return (unsigned int)((tot + bs - 1) / bs); };

  const int Et = N_EDGES + N_NODES;

  // ======================= GAT layer 1 ====================================
  gather_pack_kernel<<<nblk((size_t)N_NODES * Kp0, 256), 256, 0, stream>>>(
      node_tokens, emb1, a0f16, N_NODES, EMBD, Kp0);
  pack_bfrag_kernel<<<nblk((size_t)Np1 * Kp0, 256), 256, 0, stream>>>(
      gat1_W, w1f16, EMBD, 2000, Kp0, Np1, 0);
  gemm(a0f16, w1f16, xw1, nullptr, N_NODES, 2000, Kp0, Np1, 2000, 0);

  attn_node_kernel<<<nblk(N_NODES * HEAD1, 256), 256, 0, stream>>>(
      xw1, gat1_as, gat1_ad, asrc1, adst1, N_NODES, HEAD1, CH1);
  fill_f32_kernel<<<nblk(N_NODES * HEAD1, 256), 256, 0, stream>>>(m1, -1e30f, (size_t)N_NODES * HEAD1);
  fill_f32_kernel<<<nblk(N_NODES * HEAD1, 256), 256, 0, stream>>>(den1, 0.f, (size_t)N_NODES * HEAD1);
  fill_f32_kernel<<<nblk((size_t)N_NODES * 2000, 256), 256, 0, stream>>>(out1, 0.f, (size_t)N_NODES * 2000);
  edge_max_kernel<<<nblk(Et, 256), 256, 0, stream>>>(edge_index, N_EDGES, N_NODES, HEAD1, asrc1, adst1, m1);
  edge_denom_kernel<<<nblk(Et, 256), 256, 0, stream>>>(edge_index, N_EDGES, N_NODES, HEAD1, asrc1, adst1, m1, den1);
  edge_scatter_kernel<<<nblk((size_t)Et * 2000, 256), 256, 0, stream>>>(
      edge_index, N_EDGES, N_NODES, HEAD1, CH1, asrc1, adst1, m1, den1, xw1, out1);
  bias_elu_kernel<<<nblk((size_t)N_NODES * 2000, 256), 256, 0, stream>>>(out1, gat1_b, N_NODES, 2000);

  // ======================= GAT layer 2 ====================================
  pack_f16_kernel<<<nblk((size_t)N_NODES * Kp1, 256), 256, 0, stream>>>(
      out1, x1f16, N_NODES, 2000, Kp1);
  pack_bfrag_kernel<<<nblk((size_t)Np2 * Kp1, 256), 256, 0, stream>>>(
      gat2_W, w2f16, 2000, 1000, Kp1, Np2, 0);
  gemm(x1f16, w2f16, xw2, nullptr, N_NODES, 1000, Kp1, Np2, 1000, 0);

  attn_node_kernel<<<nblk(N_NODES * HEAD2, 256), 256, 0, stream>>>(
      xw2, gat2_as, gat2_ad, asrc2, adst2, N_NODES, HEAD2, CH2);
  fill_f32_kernel<<<nblk(N_NODES * HEAD2, 256), 256, 0, stream>>>(m2, -1e30f, (size_t)N_NODES * HEAD2);
  fill_f32_kernel<<<nblk(N_NODES * HEAD2, 256), 256, 0, stream>>>(den2, 0.f, (size_t)N_NODES * HEAD2);
  fill_f32_kernel<<<nblk((size_t)N_NODES * 1000, 256), 256, 0, stream>>>(out2, 0.f, (size_t)N_NODES * 1000);
  edge_max_kernel<<<nblk(Et, 256), 256, 0, stream>>>(edge_index, N_EDGES, N_NODES, HEAD2, asrc2, adst2, m2);
  edge_denom_kernel<<<nblk(Et, 256), 256, 0, stream>>>(edge_index, N_EDGES, N_NODES, HEAD2, asrc2, adst2, m2, den2);
  edge_scatter_kernel<<<nblk((size_t)Et * 1000, 256), 256, 0, stream>>>(
      edge_index, N_EDGES, N_NODES, HEAD2, CH2, asrc2, adst2, m2, den2, xw2, out2);
  bias_elu_kernel<<<nblk((size_t)N_NODES * 1000, 256), 256, 0, stream>>>(out2, gat2_b, N_NODES, 1000);

  // ======================= global max pool ================================
  fill_f32_kernel<<<nblk((size_t)BATCHSZ * 1000, 256), 256, 0, stream>>>(
      gpool, -1e30f, (size_t)BATCHSZ * 1000);
  pool_max_kernel<<<nblk((size_t)N_NODES * 1000, 256), 256, 0, stream>>>(
      out2, batch, gpool, N_NODES, 1000);
  pool_fix_kernel<<<nblk((size_t)BATCHSZ * 1000, 256), 256, 0, stream>>>(
      gpool, (size_t)BATCHSZ * 1000);

  // ======================= GRU stack ======================================
  gather_pack_kernel<<<nblk((size_t)BT * Kp0, 256), 256, 0, stream>>>(
      tokens, emb2, temb, BT, EMBD, Kp0);
  for (int l = 0; l < 3; ++l)
    pack_bfrag_kernel<<<nblk((size_t)NpG * Kp0, 256), 256, 0, stream>>>(
        Wih[l], wihT[l], GRUH, 300, Kp0, NpG, 1);   // src is (300,100): transpose

  for (int l = 0; l < 3; ++l) {
    const _Float16* in = (l == 0) ? temb : ysf16;
    gemm(in, wihT[l], gibuf, bih[l], BT, 300, Kp0, NpG, 300, 0);
    gru_scan_kernel<<<BATCHSZ, 320, 0, stream>>>(
        gibuf, Whh[l], bhh[l], ysA, hTbuf + (size_t)l * BATCHSZ * GRUH);
    if (l < 2)
      pack_f16_kernel<<<nblk((size_t)BT * Kp0, 256), 256, 0, stream>>>(
          ysA, ysf16, BT, GRUH, Kp0);
  }

  // ======================= MLP head =======================================
  concat_head_kernel<<<nblk((size_t)BATCHSZ * 1504, 256), 256, 0, stream>>>(gpool, hTbuf, zcat);

  pack_bfrag_kernel<<<nblk((size_t)1024 * 1504, 256), 256, 0, stream>>>(
      lin1_W, wl1, 1500, 1000, 1504, 1024, 0);
  gemm(zcat, wl1, z1, lin1_b, BATCHSZ, 1000, 1504, 1024, 1000, 1);

  pack_f16_kernel<<<nblk((size_t)BATCHSZ * 1024, 256), 256, 0, stream>>>(
      z1, z1f16, BATCHSZ, 1000, 1024);
  pack_bfrag_kernel<<<nblk((size_t)512 * 1024, 256), 256, 0, stream>>>(
      lin11_W, wl11, 1000, 500, 1024, 512, 0);
  gemm(z1f16, wl11, z2, lin11_b, BATCHSZ, 500, 1024, 512, 500, 1);

  pack_f16_kernel<<<nblk((size_t)BATCHSZ * 512, 256), 256, 0, stream>>>(
      z2, z2f16, BATCHSZ, 500, 512);
  pack_bfrag_kernel<<<nblk((size_t)32 * 512, 256), 256, 0, stream>>>(
      lin2_W, wl2, 500, 4, 512, 32, 0);
  gemm(z2f16, wl2, out, lin2_b, BATCHSZ, 4, 512, 32, 4, 1);
}
